// OnetoManyGRU_73761768341716
// MI455X (gfx1250) — compile-verified, matrix operand order checked by reading, not run on hardware
//
#include <hip/hip_runtime.h>
#include <stdint.h>

#define EMBED  1024
#define BATCH  512
#define KSTEPS 96
#define THREEH (3 * EMBED)   // 3072
#define SIXH   (6 * EMBED)   // 6144

// GEMM tiling: C[BM x BN] per workgroup, K-sliced by BK, double-buffered LDS.
#define BM 128
#define BN 128
#define BK 64
#define LDT 72                    // padded LDS k-stride (elements) -> 144B rows, conflict-free b128 reads
#define TILE_ELEMS (BM * LDT)     // 9216 elements per buffer
#define NSLICE (EMBED / BK)       // 16

typedef __bf16 v16bf __attribute__((ext_vector_type(16)));
typedef float  v8f   __attribute__((ext_vector_type(8)));

union FragU {
    v16bf f;
    uint4 q[2];
};

__device__ __forceinline__ unsigned short f2bf(float f) {
    union { float f; unsigned u; } x;
    x.f = f;
    unsigned r = x.u + 0x7FFFu + ((x.u >> 16) & 1u);  // RNE
    return (unsigned short)(r >> 16);
}

// Async global -> LDS, 16B per lane, GVS addressing (saddr base + 32-bit voffset).
__device__ __forceinline__ void async_copy_b128(uint32_t lds_addr, uint32_t goff, uint64_t gbase) {
    asm volatile("global_load_async_to_lds_b128 %0, %1, %2"
                 :: "v"(lds_addr), "v"(goff), "s"(gbase)
                 : "memory");
}

// ---------------------------------------------------------------------------
// Init: pack W_ih/W_hh into one bf16 [6144 x 1024] matrix, seed h (f32 + bf16)
// from c, and zero the x0 bf16 buffer.
// ---------------------------------------------------------------------------
__global__ void gru_init(const float* __restrict__ Wih, const float* __restrict__ Whh,
                         const float* __restrict__ c,
                         unsigned short* __restrict__ Wcat,
                         unsigned short* __restrict__ xB,
                         unsigned short* __restrict__ hB,
                         float* __restrict__ hF) {
    const size_t stride = (size_t)gridDim.x * blockDim.x;
    const size_t tid = (size_t)blockIdx.x * blockDim.x + threadIdx.x;
    const size_t NW   = (size_t)SIXH * EMBED;     // 6.29M
    const size_t HALF = (size_t)THREEH * EMBED;   // 3.15M
    for (size_t i = tid; i < NW; i += stride)
        Wcat[i] = f2bf(i < HALF ? Wih[i] : Whh[i - HALF]);
    const size_t NH = (size_t)BATCH * EMBED;
    for (size_t i = tid; i < NH; i += stride) {
        float v = c[i];
        hB[i] = f2bf(v);
        hF[i] = v;
        xB[i] = 0;  // bf16 +0.0
    }
}

// ---------------------------------------------------------------------------
// Fused-gate GEMM: G[512 x 6144] = A @ Wcat^T  (bf16 in, f32 accum).
// Columns [0,3072) use A_gi (x, zero at step 0); [3072,6144) use A_gh (h).
// 8 waves per WG; wave computes 32x64 via 2x4 WMMA 16x16x32 tiles.
// ---------------------------------------------------------------------------
__global__ __launch_bounds__(256) void gru_gemm(const unsigned short* __restrict__ Agi,
                                                const unsigned short* __restrict__ Agh,
                                                const unsigned short* __restrict__ Wcat,
                                                float* __restrict__ G) {
    __shared__ __align__(16) unsigned short shA[2][TILE_ELEMS];
    __shared__ __align__(16) unsigned short shB[2][TILE_ELEMS];

    const int tid  = threadIdx.x;
    const int wave = tid >> 5;
    const int lane = tid & 31;
    const int m0 = blockIdx.y * BM;
    const int n0 = blockIdx.x * BN;

    const unsigned short* A = (n0 < THREEH) ? Agi : Agh;
    const uint64_t aBase = (uint64_t)(uintptr_t)A;
    const uint64_t bBase = (uint64_t)(uintptr_t)Wcat;

    const int waveM = wave & 3;   // 4 waves along M (4*32 = 128)
    const int waveN = wave >> 2;  // 2 waves along N (2*64 = 128)

    v8f acc[2][4];
#pragma unroll
    for (int i = 0; i < 2; ++i)
#pragma unroll
        for (int j = 0; j < 4; ++j)
            acc[i][j] = v8f{};

    // Each wave issues 8 async b128 ops per K-slice (4 for A, 4 for B):
    // per slice the WG moves 16KB of A and 16KB of B.
    auto prefetch = [&](int buf, int k0) {
#pragma unroll
        for (int it = 0; it < 4; ++it) {
            int cid = it * 256 + tid;            // 1024 chunks of 8 elements
            int row = cid >> 3;                  // 0..127
            int kc  = (cid & 7) << 3;            // 0,8,...,56
            uint32_t goff = (uint32_t)(((m0 + row) * EMBED + k0 + kc) * 2);
            uint32_t laddr = (uint32_t)(uintptr_t)(&shA[buf][row * LDT + kc]);
            async_copy_b128(laddr, goff, aBase);
        }
#pragma unroll
        for (int it = 0; it < 4; ++it) {
            int cid = it * 256 + tid;
            int row = cid >> 3;                  // weight row j = n0 + row
            int kc  = (cid & 7) << 3;
            uint32_t goff = (uint32_t)(((n0 + row) * EMBED + k0 + kc) * 2);
            uint32_t laddr = (uint32_t)(uintptr_t)(&shB[buf][row * LDT + kc]);
            async_copy_b128(laddr, goff, bBase);
        }
    };

    prefetch(0, 0);

    for (int s = 0; s < NSLICE; ++s) {
        const int buf = s & 1;
        if (s + 1 < NSLICE) {
            prefetch(buf ^ 1, (s + 1) * BK);
            // 8 newest ops (next slice) may stay in flight; this slice is done.
            asm volatile("s_wait_asynccnt 8" ::: "memory");
        } else {
            asm volatile("s_wait_asynccnt 0" ::: "memory");
        }
        __syncthreads();

#pragma unroll
        for (int ks = 0; ks < BK; ks += 32) {
            const int kk = ks + ((lane >> 4) << 3);  // lanes 16-31 take K+8 per fragment layout
            FragU a[2];
#pragma unroll
            for (int mt = 0; mt < 2; ++mt) {
                int row = waveM * 32 + mt * 16 + (lane & 15);
                const unsigned short* p = &shA[buf][row * LDT + kk];
                a[mt].q[0] = *(const uint4*)(p);        // K 0..7  (or 8..15)
                a[mt].q[1] = *(const uint4*)(p + 16);   // K 16..23 (or 24..31)
            }
            FragU b[4];
#pragma unroll
            for (int nt = 0; nt < 4; ++nt) {
                int col = waveN * 64 + nt * 16 + (lane & 15);
                const unsigned short* p = &shB[buf][col * LDT + kk];
                b[nt].q[0] = *(const uint4*)(p);
                b[nt].q[1] = *(const uint4*)(p + 16);
            }
#pragma unroll
            for (int mt = 0; mt < 2; ++mt)
#pragma unroll
                for (int nt = 0; nt < 4; ++nt)
                    acc[mt][nt] = __builtin_amdgcn_wmma_f32_16x16x32_bf16(
                        false, a[mt].f, false, b[nt].f,
                        (short)0, acc[mt][nt], false, false);
        }
        __syncthreads();  // protect buf from being overwritten by next prefetch
    }

    // Epilogue: C/D layout — VGPR r holds M=r (lanes 0-15) / M=r+8 (lanes 16-31),
    // N = lane&15. Lanes 0-15 of a row write 64 contiguous bytes.
    const int mBase = m0 + waveM * 32 + ((lane >> 4) << 3);
    const int nBase = n0 + waveN * 64 + (lane & 15);
#pragma unroll
    for (int mt = 0; mt < 2; ++mt)
#pragma unroll
        for (int nt = 0; nt < 4; ++nt)
#pragma unroll
            for (int r = 0; r < 8; ++r) {
                int m = mBase + mt * 16 + r;
                int n = nBase + nt * 16;
                G[(size_t)m * SIXH + n] = acc[mt][nt][r];
            }
}

// ---------------------------------------------------------------------------
// Gate nonlinearity + state update + output write. One thread per (b, e).
// ---------------------------------------------------------------------------
__global__ __launch_bounds__(256) void gru_gates(const float* __restrict__ G,
                                                 const float* __restrict__ b_ih,
                                                 const float* __restrict__ b_hh,
                                                 float* __restrict__ hF,
                                                 unsigned short* __restrict__ hB,
                                                 float* __restrict__ y,
                                                 int k) {
    const int idx = blockIdx.x * 256 + threadIdx.x;  // 0 .. BATCH*EMBED-1
    const int b = idx >> 10;
    const int e = idx & (EMBED - 1);
    const float* g = G + (size_t)b * SIXH;

    float ir = g[e]             + b_ih[e];
    float iz = g[EMBED + e]     + b_ih[EMBED + e];
    float in = g[2 * EMBED + e] + b_ih[2 * EMBED + e];
    float hr = g[THREEH + e]             + b_hh[e];
    float hz = g[THREEH + EMBED + e]     + b_hh[EMBED + e];
    float hn = g[THREEH + 2 * EMBED + e] + b_hh[2 * EMBED + e];

    float r = 1.0f / (1.0f + __expf(-(ir + hr)));
    float z = 1.0f / (1.0f + __expf(-(iz + hz)));
    float n = tanhf(in + r * hn);

    float hprev = hF[idx];
    float hnew  = (1.0f - z) * n + z * hprev;

    hF[idx] = hnew;
    hB[idx] = f2bf(hnew);
    y[((size_t)b * KSTEPS + k) * EMBED + e] = hnew;
}

// ---------------------------------------------------------------------------
// Host launch: init once, then 96 dependent (GEMM, gates) pairs on `stream`.
// ---------------------------------------------------------------------------
extern "C" void kernel_launch(void* const* d_in, const int* in_sizes, int n_in,
                              void* d_out, int out_size, void* d_ws, size_t ws_size,
                              hipStream_t stream) {
    const float* c   = (const float*)d_in[0];
    // d_in[1] is K (device scalar) — fixed to KSTEPS=96 per the reference.
    const float* Wih = (const float*)d_in[2];
    const float* Whh = (const float*)d_in[3];
    const float* bih = (const float*)d_in[4];
    const float* bhh = (const float*)d_in[5];
    float* y = (float*)d_out;

    char* ws = (char*)d_ws;
    size_t off = 0;
    auto take = [&](size_t bytes) -> void* {
        off = (off + 255) & ~(size_t)255;
        void* p = ws + off;
        off += bytes;
        return p;
    };
    unsigned short* Wcat = (unsigned short*)take((size_t)SIXH * EMBED * 2);   // 12.6 MB
    unsigned short* xB   = (unsigned short*)take((size_t)BATCH * EMBED * 2);  // zeros (step-0 x)
    unsigned short* hB   = (unsigned short*)take((size_t)BATCH * EMBED * 2);  // bf16 hidden
    float*          hF   = (float*)take((size_t)BATCH * EMBED * 4);           // f32 hidden
    float*          G    = (float*)take((size_t)BATCH * SIXH * 4);            // 12.6 MB gate pre-acts
    (void)ws_size; (void)in_sizes; (void)n_in; (void)out_size;

    gru_init<<<1024, 256, 0, stream>>>(Wih, Whh, c, Wcat, xB, hB, hF);

    const dim3 ggrid(SIXH / BN, BATCH / BM);  // 48 x 4 = 192 WGs per step
    for (int k = 0; k < KSTEPS; ++k) {
        const unsigned short* Agi = (k == 0) ? xB : hB;  // x0 == 0
        gru_gemm<<<ggrid, 256, 0, stream>>>(Agi, hB, Wcat, G);
        gru_gates<<<(BATCH * EMBED) / 256, 256, 0, stream>>>(G, bih, bhh, hF, hB, y, k);
    }
}
